// GNNRNN_27307402068443
// MI455X (gfx1250) — matrix operand order repaired and checked by reading, not compiled, not run on hardware
//
#include <hip/hip_runtime.h>
#include <hip/hip_bf16.h>

#define B_ 128
#define T_ 64
#define F_ 128
#define H_ 64
#define HEADS_ 4
#define OUT_ 16
#define G3_ (3 * H_)     // 192
#define MAXDEG_ 192      // P(deg > 192) ~ 0 for E=1152 random over 128 dsts

typedef __bf16 bf16_t;
typedef __attribute__((ext_vector_type(16))) __bf16 v16bf;
typedef __attribute__((ext_vector_type(8)))  float  v8f;

// ---------------------------------------------------------------- helpers ---

__device__ __forceinline__ bf16_t f2bf(float f) {
  unsigned u = __builtin_bit_cast(unsigned, f);
  unsigned r = u + 0x7FFFu + ((u >> 16) & 1u);     // round-to-nearest-even
  unsigned short h = (unsigned short)(r >> 16);
  return __builtin_bit_cast(bf16_t, h);
}

// A fragment: 16x32 bf16 tile, rows m0..m0+15, cols k0..k0+31 from row-major LDS.
// CDNA5 ISA 7.12.2: lanes 0-15 M=lane, K in {0..7}U{16..23}; lanes 16-31 same M,
// K offset +8.
__device__ __forceinline__ v16bf load_frag_a(const bf16_t* lds, int lda,
                                             int m0, int k0, int lane) {
  v16bf a;
  int m  = m0 + (lane & 15);
  int kb = (lane < 16) ? 0 : 8;
  const bf16_t* row = lds + m * lda + k0;
#pragma unroll
  for (int i = 0; i < 16; ++i) {
    int v = i >> 1;
    int k = (v < 4) ? ((v << 1) | (i & 1)) : (16 + (((v - 4) << 1) | (i & 1)));
    a[i] = row[kb + k];
  }
  return a;
}

// B fragment: 32x16 where B(k,n) = W[n][k], W row-major [n][k] in LDS.
// Lanes 0-15: N=lane, K in {0..7}U{16..23}; lanes 16-31: N=lane-16, K offset +8.
__device__ __forceinline__ v16bf load_frag_b(const bf16_t* lds, int ldw,
                                             int n0, int k0, int lane) {
  v16bf b;
  int n  = n0 + (lane & 15);
  int kb = (lane < 16) ? 0 : 8;
  const bf16_t* row = lds + n * ldw + k0 + kb;
#pragma unroll
  for (int i = 0; i < 16; ++i) b[i] = row[(i < 8) ? i : (i + 8)];
  return b;
}

// ------------------------------------------------------------------- init ---

__global__ void init_kernel(int* bar_ctr) { *bar_ctr = 0; }

// ------------------------------------------------ persistent fused kernel ---
// One block per feature f, resident for all T steps. W_hh[f], gat_W, gate
// params, edge list and recurrent state h[f] live in LDS for the whole scan.
// Cross-block data (z, el) is double-buffered in global memory; one grid-wide
// barrier per step separates produce (GEMM phases) from consume (aggregate).

__global__ __launch_bounds__(256) void fused_scan_kernel(
    const float* __restrict__ hist,
    const int* __restrict__ srcE, const int* __restrict__ dstE, int E,
    const float* __restrict__ W_ih, const float* __restrict__ b_ih,
    const float* __restrict__ W_hh, const float* __restrict__ b_hh,
    const float* __restrict__ gat_W, const float* __restrict__ attn_l,
    const float* __restrict__ attn_r, const float* __restrict__ gat_bias,
    const float* __restrict__ W_dec, const float* __restrict__ b_dec,
    float* __restrict__ z_g,   // [2][F*B*H]   double buffer
    float* __restrict__ el_g,  // [2][F*B*HEADS]
    int* __restrict__ bar_ctr,
    float* __restrict__ ans_out, float* __restrict__ hiddens_out) {
  const int f   = blockIdx.x;
  const int tid = threadIdx.x;

  __shared__ bf16_t wl[G3_][H_ + 8];       // 27.6 KB  W_hh[f] bf16 (resident)
  __shared__ bf16_t gw[H_][H_ + 8];        //  9.2 KB  gat_W bf16  (resident)
  __shared__ float  hf[B_][H_];            // 32   KB  h[f] f32    (resident)
  __shared__ bf16_t hb[B_][H_ + 8];        // 18.4 KB  h[f] bf16 (GEMM A)
  __shared__ bf16_t hidb[B_][H_ + 8];      // 18.4 KB  hid bf16 (GEMM A)
  __shared__ float  er_s[B_ * HEADS_];     //  2 KB
  __shared__ float  elacc[B_][HEADS_], eracc[B_][HEADS_];
  __shared__ float  ansacc[B_];
  __shared__ float  wih_s[G3_], bih_s[G3_], bhh_s[G3_], wdec_s[H_];
  __shared__ float  al_s[H_], ar_s[H_], gb_s[H_];
  __shared__ int    col_s[MAXDEG_];
  __shared__ int    deg_s;

  // ---------------- one-time setup ----------------
  for (int i = tid; i < G3_ * H_; i += 256)
    wl[i >> 6][i & 63] = f2bf(W_hh[f * G3_ * H_ + i]);
  for (int i = tid; i < H_ * H_; i += 256)
    gw[i >> 6][i & 63] = f2bf(gat_W[i]);
  for (int i = tid; i < B_ * H_; i += 256)
    hf[i >> 6][i & 63] = 0.0f;                       // h0 = 0
  for (int i = tid; i < G3_; i += 256) {
    wih_s[i] = W_ih[f * G3_ + i];
    bih_s[i] = b_ih[f * G3_ + i];
    bhh_s[i] = b_hh[f * G3_ + i];
  }
  for (int i = tid; i < H_; i += 256) {
    wdec_s[i] = W_dec[f * H_ + i];
    al_s[i]   = attn_l[i];
    ar_s[i]   = attn_r[i];
    gb_s[i]   = gat_bias[i];
  }
  if (tid == 0) deg_s = 0;
  __syncthreads();
  for (int e = tid; e < E; e += 256) {               // this block's in-edges
    if (dstE[e] == f) {
      int pos = atomicAdd(&deg_s, 1);
      if (pos < MAXDEG_) col_s[pos] = srcE[e];
    }
  }
  __syncthreads();
  const int deg = (deg_s < MAXDEG_) ? deg_s : MAXDEG_;
  const float bdec = b_dec[f];

  const int lane = tid & 31;
  const int wave = tid >> 5;
  const int m0 = wave * 16;
  const int n  = lane & 15;

  // ---------------- scan over time ----------------
  for (int t = 0; t < T_; ++t) {
    const int par = t & 1;
    float* zb  = z_g  + (size_t)par * F_ * B_ * H_;
    float* elb = el_g + (size_t)par * F_ * B_ * HEADS_;

    // ---- phase 1: GRU  (gh = h x W_hh^T via WMMA, gates in regs) ----
    for (int i = tid; i < B_ * H_; i += 256)
      hb[i >> 6][i & 63] = f2bf(hf[i >> 6][i & 63]);
    if (tid < B_) ansacc[tid] = 0.0f;
    __syncthreads();

    v8f acc[12];
#pragma unroll
    for (int j = 0; j < 12; ++j)
#pragma unroll
      for (int r = 0; r < 8; ++r) acc[j][r] = 0.0f;
#pragma unroll
    for (int kk = 0; kk < H_; kk += 32) {
      v16bf a = load_frag_a(&hb[0][0], H_ + 8, m0, kk, lane);
#pragma unroll
      for (int j = 0; j < 12; ++j) {
        v16bf b = load_frag_b(&wl[0][0], H_ + 8, j * 16, kk, lane);
        acc[j] = __builtin_amdgcn_wmma_f32_16x16x32_bf16(
            false, a, false, b, (short)0, acc[j], false, false);
      }
    }

#pragma unroll
    for (int r = 0; r < 8; ++r) {
      int m = m0 + ((lane < 16) ? r : (8 + r));       // batch row
      float x = hist[(m * T_ + t) * F_ + f];
      if (t + 1 < T_)
        __builtin_prefetch(&hist[(m * T_ + t + 1) * F_ + f], 0, 0);
      float dec = 0.0f;
#pragma unroll
      for (int j = 0; j < 4; ++j) {
        int g = j * 16 + n;
        float hr = acc[j][r]     + bhh_s[g];
        float hz = acc[j + 4][r] + bhh_s[H_ + g];
        float hn = acc[j + 8][r] + bhh_s[2 * H_ + g];
        float ir  = x * wih_s[g]          + bih_s[g];
        float iz  = x * wih_s[H_ + g]     + bih_s[H_ + g];
        float in_ = x * wih_s[2 * H_ + g] + bih_s[2 * H_ + g];
        float rg = 1.0f / (1.0f + __expf(-(ir + hr)));
        float zg = 1.0f / (1.0f + __expf(-(iz + hz)));
        float ng = tanhf(in_ + rg * hn);
        float hid = (1.0f - zg) * ng + zg * hf[m][g];
        hidb[m][g] = f2bf(hid);
        dec += hid * wdec_s[g];
      }
      atomicAdd(&ansacc[m], dec);
    }
    __syncthreads();
    if (tid < B_)
      ans_out[(tid * T_ + t) * F_ + f] = ansacc[tid] + bdec;
    for (int i = tid; i < B_ * HEADS_; i += 256) {
      elacc[i >> 2][i & 3] = 0.0f;
      eracc[i >> 2][i & 3] = 0.0f;
    }
    __syncthreads();

    // ---- phase 2: GAT projection z = hid x gat_W^T via WMMA ----
    v8f zacc[HEADS_];
#pragma unroll
    for (int j = 0; j < HEADS_; ++j)
#pragma unroll
      for (int r = 0; r < 8; ++r) zacc[j][r] = 0.0f;
#pragma unroll
    for (int kk = 0; kk < H_; kk += 32) {
      v16bf a = load_frag_a(&hidb[0][0], H_ + 8, m0, kk, lane);
#pragma unroll
      for (int j = 0; j < HEADS_; ++j) {              // N tile j == head j
        v16bf b = load_frag_b(&gw[0][0], H_ + 8, j * 16, kk, lane);
        zacc[j] = __builtin_amdgcn_wmma_f32_16x16x32_bf16(
            false, a, false, b, (short)0, zacc[j], false, false);
      }
    }
#pragma unroll
    for (int j = 0; j < HEADS_; ++j) {
      float al = al_s[j * OUT_ + n];
      float ar = ar_s[j * OUT_ + n];
#pragma unroll
      for (int r = 0; r < 8; ++r) {
        int m = m0 + ((lane < 16) ? r : (8 + r));
        float zv = zacc[j][r];
        zb[(f * B_ + m) * H_ + j * OUT_ + n] = zv;
        atomicAdd(&elacc[m][j], zv * al);
        atomicAdd(&eracc[m][j], zv * ar);
      }
    }
    __syncthreads();
    for (int i = tid; i < B_ * HEADS_; i += 256) {
      elb[f * B_ * HEADS_ + i] = elacc[i >> 2][i & 3];
      er_s[i] = eracc[i >> 2][i & 3];
    }

    // ---- phase 3: grid-wide barrier (publish z/el, then gather) ----
    if (tid == 0) {
      __threadfence();
      __hip_atomic_fetch_add(bar_ctr, 1, __ATOMIC_RELEASE,
                             __HIP_MEMORY_SCOPE_AGENT);
      const int target = (int)gridDim.x * (t + 1);
      while (__hip_atomic_load(bar_ctr, __ATOMIC_ACQUIRE,
                               __HIP_MEMORY_SCOPE_AGENT) < target)
        __builtin_amdgcn_s_sleep(1);
    }
    __syncthreads();

    // ---- phase 4: edge softmax + aggregate -> new h[f] ----
#pragma unroll
    for (int pp = 0; pp < 2; ++pp) {
      int p = tid + pp * 256;                         // (b, head), b-major
      int b = p >> 2, head = p & 3;
      float erv = er_s[p];

      float mx = -1e30f;
      for (int e = 0; e < deg; ++e) {
        int s = col_s[e];
        float v = elb[(s * B_ + b) * HEADS_ + head] + erv;
        v = (v > 0.0f) ? v : 0.2f * v;
        mx = fmaxf(mx, v);
      }
      float ssum = 0.0f;
      for (int e = 0; e < deg; ++e) {
        int s = col_s[e];
        float v = elb[(s * B_ + b) * HEADS_ + head] + erv;
        v = (v > 0.0f) ? v : 0.2f * v;
        ssum += __expf(v - mx);
      }
      float inv = 1.0f / ssum;

      float macc[OUT_];
#pragma unroll
      for (int o = 0; o < OUT_; ++o) macc[o] = 0.0f;
      for (int e = 0; e < deg; ++e) {
        int s = col_s[e];
        float v = elb[(s * B_ + b) * HEADS_ + head] + erv;
        v = (v > 0.0f) ? v : 0.2f * v;
        float a = __expf(v - mx) * inv;
        const float* zp = &zb[(s * B_ + b) * H_ + head * OUT_];
#pragma unroll
        for (int o = 0; o < OUT_; ++o) macc[o] += a * zp[o];
      }
#pragma unroll
      for (int o = 0; o < OUT_; ++o) {
        int g = head * OUT_ + o;
        float val = macc[o] + gb_s[g];
        hf[b][g] = val;                               // resident carry
        hiddens_out[((size_t)(b * T_ + t) * F_ + f) * H_ + g] = val;
      }
    }
    __syncthreads();                                  // hf complete for t+1
  }
}

// ------------------------------------------------------------------ driver --

extern "C" void kernel_launch(void* const* d_in, const int* in_sizes, int n_in,
                              void* d_out, int out_size, void* d_ws, size_t ws_size,
                              hipStream_t stream) {
  const float* hist   = (const float*)d_in[0];
  const int*   srcE   = (const int*)  d_in[1];
  const int*   dstE   = (const int*)  d_in[2];
  const float* W_ih   = (const float*)d_in[3];
  const float* b_ih   = (const float*)d_in[4];
  const float* W_hh   = (const float*)d_in[5];
  const float* b_hh   = (const float*)d_in[6];
  const float* gat_W  = (const float*)d_in[7];
  const float* attn_l = (const float*)d_in[8];
  const float* attn_r = (const float*)d_in[9];
  const float* gbias  = (const float*)d_in[10];
  const float* W_dec  = (const float*)d_in[11];
  const float* b_dec  = (const float*)d_in[12];
  const int E = in_sizes[1];

  float* ans_out     = (float*)d_out;                        // [B,T,F]
  float* hiddens_out = (float*)d_out + (size_t)B_ * T_ * F_; // [B,T,F,H]

  char* ws = (char*)d_ws;
  size_t off = 0;
  auto take = [&](size_t bytes) {
    size_t o = off; off = (off + bytes + 255) & ~(size_t)255; return o;
  };
  float* z_g  = (float*)(ws + take((size_t)2 * F_ * B_ * H_ * 4));      // 8 MB
  float* el_g = (float*)(ws + take((size_t)2 * F_ * B_ * HEADS_ * 4));  // 512 KB
  int*   bar  = (int*)  (ws + take(256));
  (void)ws_size; (void)n_in; (void)out_size;

  init_kernel<<<1, 1, 0, stream>>>(bar);
  fused_scan_kernel<<<F_, 256, 0, stream>>>(
      hist, srcE, dstE, E, W_ih, b_ih, W_hh, b_hh, gat_W, attn_l, attn_r,
      gbias, W_dec, b_dec, z_g, el_g, bar, ans_out, hiddens_out);
}
